// BaselineRnn_1408749273390
// MI455X (gfx1250) — matrix operand order repaired and checked by reading, not compiled, not run on hardware
//
#include <hip/hip_runtime.h>
#include <hip/hip_bf16.h>
#include <cstdint>

// ---------------------------------------------------------------------------
// LSTM (tf LSTMCell, forget_bias=1) + per-timestep dense, for MI455X (gfx1250)
// B=64, T=512, I=128, H=1024, O=128.
//   - bf16 WMMA (v_wmma_f32_16x16x32_bf16), f32 accumulate
//   - weights pre-transposed once -> BOTH A and B panels staged to LDS with
//     global_load_async_to_lds_b128 (ASYNCcnt), double-buffered
//   - fragments = contiguous ds_load_b128 pairs, zero register shuffling
//   - 512 step kernels (graph-replayed), one big WMMA GEMM for the output
// ---------------------------------------------------------------------------

typedef __attribute__((ext_vector_type(16))) __bf16 v16bf;
typedef __attribute__((ext_vector_type(8)))  __bf16 v8bf;
typedef __attribute__((ext_vector_type(8)))  float  v8f;

#define B_  64
#define T_  512
#define I_  128
#define H_  1024
#define O_  128
#define KTOT (I_ + H_)   // 1152

// fast sigmoid / tanh: v_exp_f32 + v_rcp_f32
static __device__ __forceinline__ float sigmoidf_(float x) {
    return __builtin_amdgcn_rcpf(1.0f + __expf(-x));
}
static __device__ __forceinline__ float tanhf_(float x) {
    return 2.0f * __builtin_amdgcn_rcpf(1.0f + __expf(-2.0f * x)) - 1.0f;
}

// async copy of 16 bytes: global (sbase + voff) -> LDS[ldsDst], per lane
static __device__ __forceinline__ void async_b128(uint32_t ldsDst, uint32_t voff,
                                                  uint64_t sbase) {
    asm volatile("global_load_async_to_lds_b128 %0, %1, %2"
                 :: "v"(ldsDst), "v"(voff), "s"(sbase) : "memory");
}
static __device__ __forceinline__ void wait_async0() {
    asm volatile("s_wait_asynccnt 0" ::: "memory");
}

// build v16bf WMMA fragment from two contiguous 16B LDS chunks
static __device__ __forceinline__ v16bf frag16(const __bf16* p0, const __bf16* p1) {
    v8bf lo = *(const v8bf*)p0;
    v8bf hi = *(const v8bf*)p1;
    return __builtin_shufflevector(lo, hi, 0,1,2,3,4,5,6,7,8,9,10,11,12,13,14,15);
}

// ---------------------------------------------------------------------------
// fp32 -> bf16 plain copy, 8 elements / thread
// ---------------------------------------------------------------------------
__global__ void f2bf_vec_kernel(const float* __restrict__ in,
                                __bf16* __restrict__ out, int n8) {
    int i = blockIdx.x * 256 + threadIdx.x;
    if (i < n8) {
        const float4* p = (const float4*)in + (size_t)i * 2;
        float4 a = p[0], b = p[1];
        v8bf o;
        o[0] = (__bf16)a.x; o[1] = (__bf16)a.y; o[2] = (__bf16)a.z; o[3] = (__bf16)a.w;
        o[4] = (__bf16)b.x; o[5] = (__bf16)b.y; o[6] = (__bf16)b.z; o[7] = (__bf16)b.w;
        *(v8bf*)(out + (size_t)i * 8) = o;
    }
}

// fp32 [K][N] -> bf16 transposed [N][ldDst] at column offset kOff (32x32 tiles)
__global__ __launch_bounds__(256) void transpose_f2bf_kernel(
    const float* __restrict__ src, __bf16* __restrict__ dst,
    int N, int ldDst, int kOff) {
    __shared__ __bf16 tile[32][34];
    const int n0 = blockIdx.x * 32;
    const int k0 = blockIdx.y * 32;
    const int c = threadIdx.x & 31, r8 = threadIdx.x >> 5;
    #pragma unroll
    for (int rr = 0; rr < 4; ++rr) {
        int r = r8 + rr * 8;
        tile[r][c] = (__bf16)src[(size_t)(k0 + r) * N + n0 + c];
    }
    __syncthreads();
    #pragma unroll
    for (int rr = 0; rr < 4; ++rr) {
        int r = r8 + rr * 8;
        dst[(size_t)(n0 + r) * ldDst + kOff + k0 + c] = tile[c][r];
    }
}

__global__ void init_state_kernel(float* __restrict__ c, __bf16* __restrict__ h0, int n) {
    int i = blockIdx.x * 256 + threadIdx.x;
    if (i < n) { c[i] = 0.0f; h0[i] = (__bf16)0.0f; }
}

// ---------------------------------------------------------------------------
// One LSTM timestep. Grid: 32 WGs x 256 threads (8 waves).
// WG owns 32 h-columns; wave w: batch tile rt=(w&3)*16, col tile cw=(w>>2)*16.
// Each wave holds 4 gate accumulators (i,j,f,o) of 16x16 -> cell update fully
// in registers. K = 1152 = 128 (x_t) + 1024 (h), 32 per WMMA, double-buffered.
// Wcat = [Wx;Wh] transposed: [4H][KTOT] bf16 (contiguous K per column).
// ---------------------------------------------------------------------------
__global__ __launch_bounds__(256) void lstm_step_kernel(
    const __bf16* __restrict__ Xbf,   // [B, T, I]
    const __bf16* __restrict__ Wcat,  // [4H, KTOT] (transposed weights)
    const float*  __restrict__ bias,  // [4H] gate order i, j, f, o
    const __bf16* __restrict__ hprev, // [B, H]
    __bf16*       __restrict__ hnext, // [B, H]
    float*        __restrict__ c,     // [B, H]
    __bf16*       __restrict__ hs,    // [B, T, H]
    int t)
{
    __shared__ __bf16 ldsA[2][B_ ][40];   // 64 rows x 32 K, double-buffered
    __shared__ __bf16 ldsB[2][128][40];   // 128 cols x 32 K, double-buffered

    const int tid  = threadIdx.x;
    const int wave = tid >> 5;
    const int lane = tid & 31;
    const int n0   = blockIdx.x * 32;
    const int rt   = (wave & 3) * 16;
    const int cw   = (wave >> 2) * 16;

    // --- A staging geometry: 64 rows x 4 chunks, one b128 per thread ---
    const int arow = tid >> 2, achk = tid & 3;
    const uint32_t aXrow = (uint32_t)arow * (T_ * I_ * 2);  // byte offsets
    const uint32_t aHrow = (uint32_t)arow * (H_ * 2);
    const uint64_t xBase = (uint64_t)(uintptr_t)(Xbf + (size_t)t * I_);
    const uint64_t hBase = (uint64_t)(uintptr_t)hprev;

    // --- B staging geometry: 128 cols x 4 chunks = 512 b128, 2 per thread ---
    const uint64_t wBase = (uint64_t)(uintptr_t)Wcat;
    const int bP0 = tid >> 1, bC0 = (tid & 1) * 2;          // cols 0..127, chunks {0,1},{2,3}
    // unit u covers chunk bC0+u of col bP0 -> one scheme: 2 chunks same col
    const uint32_t gcol0 = (uint32_t)((bP0 >> 5) * H_ + n0 + (bP0 & 31));
    const uint32_t bOffBase = gcol0 * (KTOT * 2);

    v8f acc[4];
    #pragma unroll
    for (int q = 0; q < 4; ++q) acc[q] = (v8f)0.0f;

    auto stage = [&](int pi, int bf) {
        const int k0 = pi * 32;
        const bool fromX = (k0 < I_);
        uint32_t aDst = (uint32_t)(uintptr_t)&ldsA[bf][arow][achk * 8];
        uint32_t voffA = fromX ? (aXrow + (uint32_t)(k0 + achk * 8) * 2)
                               : (aHrow + (uint32_t)(k0 - I_ + achk * 8) * 2);
        async_b128(aDst, voffA, fromX ? xBase : hBase);
        #pragma unroll
        for (int u = 0; u < 2; ++u) {
            int chk = bC0 + u;
            uint32_t bDst = (uint32_t)(uintptr_t)&ldsB[bf][bP0][chk * 8];
            async_b128(bDst, bOffBase + (uint32_t)(k0 + chk * 8) * 2, wBase);
        }
    };

    const int half = lane >> 4;
    const int lr   = rt + (lane & 15);
    auto compute = [&](int bf) {
        v16bf af = frag16(&ldsA[bf][lr][half * 8], &ldsA[bf][lr][16 + half * 8]);
        #pragma unroll
        for (int q = 0; q < 4; ++q) {
            const int p = q * 32 + cw + (lane & 15);
            v16bf bfr = frag16(&ldsB[bf][p][half * 8], &ldsB[bf][p][16 + half * 8]);
            acc[q] = __builtin_amdgcn_wmma_f32_16x16x32_bf16(
                false, af, false, bfr, (short)0, acc[q], false, false);
        }
    };

    constexpr int NP = KTOT / 32;   // 36 panels
    stage(0, 0);
    wait_async0();
    __syncthreads();
    for (int pi = 0; pi < NP; ++pi) {
        const int cur = pi & 1;
        if (pi + 1 < NP) stage(pi + 1, cur ^ 1);   // overlap next panel with WMMA
        compute(cur);
        if (pi + 1 < NP) wait_async0();
        __syncthreads();
    }

    // ---- fused LSTM cell update (all 4 gates local to this wave) ----
    {
        const int mhalf = lane >> 4, n = lane & 15;
        const int col = n0 + cw + n;
        const float bi = bias[0 * H_ + col];
        const float bj = bias[1 * H_ + col];
        const float bf = bias[2 * H_ + col];
        const float bo = bias[3 * H_ + col];
        #pragma unroll
        for (int v = 0; v < 8; ++v) {
            int m = rt + mhalf * 8 + v;
            float si = sigmoidf_(acc[0][v] + bi);
            float tj = tanhf_(acc[1][v] + bj);
            float sf = sigmoidf_(acc[2][v] + bf + 1.0f);   // forget_bias = 1.0
            float so = sigmoidf_(acc[3][v] + bo);
            int sidx = m * H_ + col;
            float cnew = sf * c[sidx] + si * tj;
            float hnew = so * tanhf_(cnew);
            c[sidx] = cnew;
            __bf16 hb = (__bf16)hnew;
            hnext[sidx] = hb;
            hs[(uint32_t)(m * T_ + t) * H_ + col] = hb;
        }
    }
}

// ---------------------------------------------------------------------------
// out[B*T, O] = hs[B*T, H] @ Wd[H, O] + bd   (bf16 in, fp32 out)
// Wdt = Wd transposed: [O][H]. Grid: 256 WGs x 128 rows; wave w: rows
// [w*16,+16), all 128 cols (8 accs). Both panels async, double-buffered.
// ---------------------------------------------------------------------------
__global__ __launch_bounds__(256) void out_gemm_kernel(
    const __bf16* __restrict__ hs,    // [B*T, H]
    const __bf16* __restrict__ Wdt,   // [O, H] (transposed)
    const float*  __restrict__ bd,    // [O]
    float*        __restrict__ out)   // [B*T, O]
{
    __shared__ __bf16 ldsA[2][128][40];
    __shared__ __bf16 ldsB[2][128][40];

    const int tid  = threadIdx.x;
    const int wave = tid >> 5;
    const int lane = tid & 31;
    const uint32_t row0 = (uint32_t)blockIdx.x * 128;

    const uint64_t aBase = (uint64_t)(uintptr_t)(hs + (size_t)row0 * H_);
    const uint64_t bBase = (uint64_t)(uintptr_t)Wdt;
    const int bP0 = tid >> 1, bC0 = (tid & 1) * 2;
    const uint32_t bOffBase = (uint32_t)bP0 * (H_ * 2);

    v8f acc[8];
    #pragma unroll
    for (int q = 0; q < 8; ++q) acc[q] = (v8f)0.0f;

    auto stage = [&](int pi, int bf) {
        const int k0 = pi * 32;
        #pragma unroll
        for (int u = 0; u < 2; ++u) {            // A: 128 rows x 4 chunks
            int id = tid + u * 256;
            int r = id >> 2, chk = id & 3;
            uint32_t dst = (uint32_t)(uintptr_t)&ldsA[bf][r][chk * 8];
            async_b128(dst, ((uint32_t)r * H_ + (uint32_t)(k0 + chk * 8)) * 2, aBase);
        }
        #pragma unroll
        for (int u = 0; u < 2; ++u) {            // B: 128 cols x 4 chunks
            int chk = bC0 + u;
            uint32_t dst = (uint32_t)(uintptr_t)&ldsB[bf][bP0][chk * 8];
            async_b128(dst, bOffBase + (uint32_t)(k0 + chk * 8) * 2, bBase);
        }
    };

    const int half = lane >> 4;
    const int lr   = wave * 16 + (lane & 15);
    auto compute = [&](int bf) {
        v16bf af = frag16(&ldsA[bf][lr][half * 8], &ldsA[bf][lr][16 + half * 8]);
        #pragma unroll
        for (int nt = 0; nt < 8; ++nt) {
            const int p = nt * 16 + (lane & 15);
            v16bf bfr = frag16(&ldsB[bf][p][half * 8], &ldsB[bf][p][16 + half * 8]);
            acc[nt] = __builtin_amdgcn_wmma_f32_16x16x32_bf16(
                false, af, false, bfr, (short)0, acc[nt], false, false);
        }
    };

    constexpr int NP = H_ / 32;     // 32 panels
    stage(0, 0);
    wait_async0();
    __syncthreads();
    for (int pi = 0; pi < NP; ++pi) {
        const int cur = pi & 1;
        if (pi + 1 < NP) stage(pi + 1, cur ^ 1);
        compute(cur);
        if (pi + 1 < NP) wait_async0();
        __syncthreads();
    }

    {
        const int mhalf = lane >> 4, n = lane & 15;
        #pragma unroll
        for (int nt = 0; nt < 8; ++nt) {
            int col = nt * 16 + n;
            float bdc = bd[col];
            #pragma unroll
            for (int v = 0; v < 8; ++v) {
                int m = wave * 16 + mhalf * 8 + v;
                out[(row0 + m) * O_ + col] = acc[nt][v] + bdc;
            }
        }
    }
}

// ---------------------------------------------------------------------------
extern "C" void kernel_launch(void* const* d_in, const int* in_sizes, int n_in,
                              void* d_out, int out_size, void* d_ws, size_t ws_size,
                              hipStream_t stream) {
    (void)in_sizes; (void)n_in; (void)out_size; (void)ws_size;
    const float* X  = (const float*)d_in[0];
    const float* Wx = (const float*)d_in[1];
    const float* Wh = (const float*)d_in[2];
    const float* b  = (const float*)d_in[3];
    const float* Wd = (const float*)d_in[4];
    const float* bd = (const float*)d_in[5];

    const size_t nX   = (size_t)B_ * T_ * I_;     // 4,194,304
    const size_t nCat = (size_t)(4 * H_) * KTOT;  // 4,718,592  ([4H][KTOT])
    const size_t nWdT = (size_t)O_ * H_;          //   131,072  ([O][H])
    const size_t nHS  = (size_t)B_ * T_ * H_;     // 33,554,432
    const size_t nSt  = (size_t)B_ * H_;          //    65,536

    char* wsp = (char*)d_ws;
    size_t off = 0;
    auto take = [&](size_t bytes) -> char* {
        char* q = wsp + off;
        off += (bytes + 255) & ~(size_t)255;
        return q;
    };
    __bf16* Xbf  = (__bf16*)take(nX   * 2);
    __bf16* Wcat = (__bf16*)take(nCat * 2);
    __bf16* Wdt  = (__bf16*)take(nWdT * 2);
    __bf16* hsbf = (__bf16*)take(nHS  * 2);
    __bf16* hbuf = (__bf16*)take(2 * nSt * 2);  // ping-pong h state
    float*  cbuf = (float*) take(nSt * 4);
    // total ~84 MB of workspace

    f2bf_vec_kernel<<<(unsigned)((nX / 8 + 255) / 256), 256, 0, stream>>>(
        X, Xbf, (int)(nX / 8));
    // Wx [128,4096] -> Wcat[:, 0:128);  Wh [1024,4096] -> Wcat[:, 128:1152)
    transpose_f2bf_kernel<<<dim3(4 * H_ / 32, I_ / 32), 256, 0, stream>>>(
        Wx, Wcat, 4 * H_, KTOT, 0);
    transpose_f2bf_kernel<<<dim3(4 * H_ / 32, H_ / 32), 256, 0, stream>>>(
        Wh, Wcat, 4 * H_, KTOT, I_);
    // Wd [1024,128] -> Wdt [128,1024]
    transpose_f2bf_kernel<<<dim3(O_ / 32, H_ / 32), 256, 0, stream>>>(
        Wd, Wdt, O_, H_, 0);
    init_state_kernel<<<(unsigned)((nSt + 255) / 256), 256, 0, stream>>>(
        cbuf, hbuf, (int)nSt);

    for (int t = 0; t < T_; ++t) {
        const __bf16* hp = hbuf + (size_t)(t & 1) * nSt;
        __bf16*       hn = hbuf + (size_t)((t + 1) & 1) * nSt;
        lstm_step_kernel<<<H_ / 32, 256, 0, stream>>>(
            Xbf, Wcat, b, hp, hn, cbuf, hsbf, t);
    }

    out_gemm_kernel<<<(B_ * T_) / 128, 256, 0, stream>>>(
        hsbf, Wdt, bd, (float*)d_out);
}
